// LecGNN_87737591923320
// MI455X (gfx1250) — compile-verified, matrix operand order checked by reading, not run on hardware
//
#include <hip/hip_runtime.h>

#define NNODES 50000
#define NEDGES 800000
#define ESL    (NEDGES + NNODES)   // edges + self loops
#define INDIM  256
#define HIDDIM 64
#define NGRAPH 16
#define NEGSL  0.2f

typedef __attribute__((ext_vector_type(16))) _Float16 v16h;
typedef __attribute__((ext_vector_type(8)))  float    v8f;

// ---------------------------------------------------------------------------
// WMMA fragment loader: row-major f32 source, convert to f16 in registers.
// CDNA5 16-bit A/B layout (ISA 7.12.2): lanes 0-15 hold K = k0+[0..7] (halves
// 0..7) and k0+[16..23] (halves 8..15); lanes 16-31 hold k0+[8..15], k0+[24..31].
// ---------------------------------------------------------------------------
static __device__ __forceinline__ v16h frag_f32(const float* __restrict__ rowp,
                                                int k0, int lane) {
  const float* q = rowp + k0 + ((lane >> 4) << 3);
  v16h f;
#pragma unroll
  for (int i = 0; i < 8; ++i) f[i] = (_Float16)q[i];
#pragma unroll
  for (int i = 0; i < 8; ++i) f[8 + i] = (_Float16)q[16 + i];
  return f;
}

// ---------------------------------------------------------------------------
// Tiled GEMM: out[row,col] = act( sum_k A1[row,k]*W1[col,k] (+ A2*W2) + bias )
// A: [M,K] row-major f32, W: [ncols,K] row-major f32 (i.e. computes A @ W^T).
// One wave per 16x16 output tile; K stepped by 32 with v_wmma_f32_16x16x32_f16.
// ---------------------------------------------------------------------------
__global__ __launch_bounds__(32) void k_gemm16(
    const float* __restrict__ A1, const float* __restrict__ W1,
    const float* __restrict__ A2, const float* __restrict__ W2,
    const float* __restrict__ bias, int do_relu,
    float* __restrict__ out, int K, int ncols)
{
  const int lane = threadIdx.x;
  const int row0 = blockIdx.x << 4;
  const int col0 = blockIdx.y << 4;
  const int l15  = lane & 15;

  const float* a1p = A1 + (size_t)(row0 + l15) * K;
  const float* w1p = W1 + (size_t)(col0 + l15) * K;
  const float* a2p = A2 ? A2 + (size_t)(row0 + l15) * K : nullptr;
  const float* w2p = A2 ? W2 + (size_t)(col0 + l15) * K : nullptr;

  v8f acc = {};
  for (int k0 = 0; k0 < K; k0 += 32) {
    v16h a = frag_f32(a1p, k0, lane);
    v16h b = frag_f32(w1p, k0, lane);
    acc = __builtin_amdgcn_wmma_f32_16x16x32_f16(false, a, false, b,
                                                 (short)0, acc, false, false);
    if (a2p) {                                  // uniform branch (SGPR ptr)
      v16h a2 = frag_f32(a2p, k0, lane);
      v16h b2 = frag_f32(w2p, k0, lane);
      acc = __builtin_amdgcn_wmma_f32_16x16x32_f16(false, a2, false, b2,
                                                   (short)0, acc, false, false);
    }
  }

  // C/D layout: VGPR i -> row0 + i + 8*(lane>=16), col = col0 + (lane&15)
  const int half = lane >> 4;
  const int col  = col0 + l15;
  const float bv = bias ? bias[col] : 0.0f;
#pragma unroll
  for (int i = 0; i < 8; ++i) {
    const int row = row0 + i + (half << 3);
    float v = acc[i] + bv;
    if (do_relu) v = fmaxf(v, 0.0f);
    out[(size_t)row * ncols + col] = v;
  }
}

// ---------------------------------------------------------------------------
// Generic fill
// ---------------------------------------------------------------------------
__global__ void k_fill(float* __restrict__ p, float v, long long n) {
  long long i = (long long)blockIdx.x * blockDim.x + threadIdx.x;
  if (i < n) p[i] = v;
}

// ---------------------------------------------------------------------------
// SAGE stage: degree / edge-attr-sum scatter, feature scatter, in-place mean
// ---------------------------------------------------------------------------
__global__ void k_deg(const int* __restrict__ ei, const float* __restrict__ ea,
                      float* __restrict__ deg, float* __restrict__ esum) {
  int e = blockIdx.x * blockDim.x + threadIdx.x;
  if (e >= NEDGES) return;
  int d = ei[NEDGES + e];
  atomicAdd(&deg[d], 1.0f);
  atomicAdd(&esum[d], ea[e]);
}

__global__ void k_scatter_feat(const float* __restrict__ x,
                               const int* __restrict__ ei,
                               float* __restrict__ agg) {
  long long idx = (long long)blockIdx.x * blockDim.x + threadIdx.x;
  if (idx >= (long long)NEDGES * 64) return;
  int e  = (int)(idx >> 6);
  int c4 = ((int)idx & 63) << 2;
  int s = ei[e], d = ei[NEDGES + e];
  float4 v = *(const float4*)(x + (size_t)s * INDIM + c4);
  float* p = agg + (size_t)d * INDIM + c4;
  atomicAdd(p + 0, v.x); atomicAdd(p + 1, v.y);
  atomicAdd(p + 2, v.z); atomicAdd(p + 3, v.w);
}

__global__ void k_mean(float* __restrict__ agg, const float* __restrict__ deg,
                       const float* __restrict__ esum, float* __restrict__ mean_e) {
  int idx = blockIdx.x * blockDim.x + threadIdx.x;
  if (idx >= NNODES * 64) return;
  int n = idx >> 6;
  int c4 = (idx & 63) << 2;
  float inv = 1.0f / fmaxf(deg[n], 1.0f);
  float* p = agg + (size_t)n * INDIM + c4;
  p[0] *= inv; p[1] *= inv; p[2] *= inv; p[3] *= inv;
  if ((idx & 63) == 0) mean_e[n] = esum[n] * inv;
}

// ---------------------------------------------------------------------------
// GAT stage kernels
// ---------------------------------------------------------------------------
__global__ void k_wa(const float* __restrict__ We, const float* __restrict__ ae,
                     float* __restrict__ wa, int H) {
  int h = threadIdx.x;
  if (h >= H) return;
  float s = 0.0f;
  for (int c = 0; c < HIDDIM; ++c) s += We[h * HIDDIM + c] * ae[h * HIDDIM + c];
  wa[h] = s;
}

__global__ void k_scores(const float* __restrict__ hh, const float* __restrict__ as,
                         const float* __restrict__ ad, float* __restrict__ ss,
                         float* __restrict__ sd, int H) {
  int idx = blockIdx.x * blockDim.x + threadIdx.x;
  if (idx >= NNODES * H) return;
  int n = idx / H, h = idx % H;
  const float* p = hh + (size_t)n * H * HIDDIM + h * HIDDIM;
  float a = 0.0f, b = 0.0f;
  for (int c = 0; c < HIDDIM; ++c) {
    a += p[c] * as[h * HIDDIM + c];
    b += p[c] * ad[h * HIDDIM + c];
  }
  ss[idx] = a; sd[idx] = b;
}

static __device__ __forceinline__ void atomicMaxF(float* addr, float val) {
  if (val >= 0.0f) atomicMax((int*)addr, __float_as_int(val));
  else             atomicMin((unsigned int*)addr, __float_as_uint(val));
}

__global__ void k_alpha(const int* __restrict__ ei, const float* __restrict__ ea,
                        const float* __restrict__ mean_e,
                        const float* __restrict__ ss, const float* __restrict__ sd,
                        const float* __restrict__ wa,
                        float* __restrict__ alpha, float* __restrict__ m, int H) {
  long long idx = (long long)blockIdx.x * blockDim.x + threadIdx.x;
  if (idx >= (long long)ESL * H) return;
  int h = (int)(idx % H);
  long long e = idx / H;
  int s, d; float at;
  if (e < NEDGES) { s = ei[e]; d = ei[NEDGES + e]; at = ea[e]; }
  else            { s = d = (int)(e - NEDGES);     at = mean_e[d]; }
  float a = ss[(size_t)s * H + h] + sd[(size_t)d * H + h] + at * wa[h];
  a = (a > 0.0f) ? a : NEGSL * a;              // leaky_relu(0.2)
  alpha[idx] = a;
  atomicMaxF(&m[(size_t)d * H + h], a);        // segment max per dst/head
}

__global__ void k_expden(const int* __restrict__ ei, float* __restrict__ alpha,
                         const float* __restrict__ m, float* __restrict__ den, int H) {
  long long idx = (long long)blockIdx.x * blockDim.x + threadIdx.x;
  if (idx >= (long long)ESL * H) return;
  int h = (int)(idx % H);
  long long e = idx / H;
  int d = (e < NEDGES) ? ei[NEDGES + e] : (int)(e - NEDGES);
  float num = __expf(alpha[idx] - m[(size_t)d * H + h]);
  alpha[idx] = num;                            // reuse buffer: now holds numerator
  atomicAdd(&den[(size_t)d * H + h], num);
}

__global__ void k_aggmsg(const int* __restrict__ ei, const float* __restrict__ num,
                         const float* __restrict__ den, const float* __restrict__ hh,
                         float* __restrict__ outacc, int H) {
  long long idx = (long long)blockIdx.x * blockDim.x + threadIdx.x;
  long long tot = (long long)ESL * H * 16;
  if (idx >= tot) return;
  int r = (int)(idx % (H * 16));
  long long e = idx / (H * 16);
  int h  = r >> 4;
  int c4 = (r & 15) << 2;
  int s, d;
  if (e < NEDGES) { s = ei[e]; d = ei[NEDGES + e]; }
  else            { s = d = (int)(e - NEDGES); }
  float coef = num[e * H + h] / den[(size_t)d * H + h];
  const float* hp = hh + (size_t)s * (H * HIDDIM) + h * HIDDIM + c4;
  float* op = outacc + (size_t)d * HIDDIM + c4;
  atomicAdd(op + 0, hp[0] * coef); atomicAdd(op + 1, hp[1] * coef);
  atomicAdd(op + 2, hp[2] * coef); atomicAdd(op + 3, hp[3] * coef);
}

__global__ void k_finish(const float* __restrict__ outacc, const float* __restrict__ b,
                         float* __restrict__ hout, float invH) {
  int idx = blockIdx.x * blockDim.x + threadIdx.x;
  if (idx >= NNODES * HIDDIM) return;
  float v = outacc[idx] * invH + b[idx & (HIDDIM - 1)];
  hout[idx] = fmaxf(v, 0.0f);
}

// ---------------------------------------------------------------------------
// Readout: masked mean pool per graph (types 1 and 2), then 2-layer MLP
// ---------------------------------------------------------------------------
__global__ void k_pool(const float* __restrict__ h, const int* __restrict__ nt,
                       const int* __restrict__ batch, float* __restrict__ pool,
                       float* __restrict__ cnt) {
  int idx = blockIdx.x * blockDim.x + threadIdx.x;
  if (idx >= NNODES * 16) return;
  int n  = idx >> 4;
  int c4 = (idx & 15) << 2;
  int t = nt[n];
  if (t != 1 && t != 2) return;
  int g = batch[n];
  float* p = pool + (size_t)(t - 1) * NGRAPH * HIDDIM + (size_t)g * HIDDIM + c4;
  const float* hp = h + (size_t)n * HIDDIM + c4;
  atomicAdd(p + 0, hp[0]); atomicAdd(p + 1, hp[1]);
  atomicAdd(p + 2, hp[2]); atomicAdd(p + 3, hp[3]);
  if (c4 == 0) atomicAdd(&cnt[(t - 1) * NGRAPH + g], 1.0f);
}

__global__ void k_mlp(const float* __restrict__ pool, const float* __restrict__ cnt,
                      const float* __restrict__ w1, const float* __restrict__ b1,
                      const float* __restrict__ w2, const float* __restrict__ b2,
                      float* __restrict__ out) {
  int g = blockIdx.x;            // 16 graphs
  int j = threadIdx.x;           // 64 hidden units
  __shared__ float hid[64];
  float c1 = fmaxf(cnt[g], 1.0f), c2 = fmaxf(cnt[NGRAPH + g], 1.0f);
  float s = b1[j];
  for (int k = 0; k < HIDDIM; ++k)
    s += (pool[(size_t)g * HIDDIM + k] / c1) * w1[j * 128 + k];
  for (int k = 0; k < HIDDIM; ++k)
    s += (pool[(size_t)NGRAPH * HIDDIM + (size_t)g * HIDDIM + k] / c2) * w1[j * 128 + 64 + k];
  hid[j] = fmaxf(s, 0.0f);
  __syncthreads();
  if (j == 0) {
    float o = b2[0];
    for (int k = 0; k < 64; ++k) o += hid[k] * w2[k];
    out[g] = o;
  }
}

// ---------------------------------------------------------------------------
// Host orchestration
// ---------------------------------------------------------------------------
static inline int cdiv_i(long long a, long long b) { return (int)((a + b - 1) / b); }

extern "C" void kernel_launch(void* const* d_in, const int* in_sizes, int n_in,
                              void* d_out, int out_size, void* d_ws, size_t ws_size,
                              hipStream_t stream) {
  (void)in_sizes; (void)n_in; (void)out_size; (void)ws_size;
  const float* x       = (const float*)d_in[0];
  const int*   ei      = (const int*)  d_in[1];
  const int*   ntype   = (const int*)  d_in[2];
  const float* eattr   = (const float*)d_in[3];
  const int*   batch   = (const int*)  d_in[4];
  const float* sage_wl = (const float*)d_in[5];
  const float* sage_bl = (const float*)d_in[6];
  const float* sage_wr = (const float*)d_in[7];
  const float* mlp_w1  = (const float*)d_in[8];
  const float* mlp_b1  = (const float*)d_in[9];
  const float* mlp_w2  = (const float*)d_in[10];
  const float* mlp_b2  = (const float*)d_in[11];
  float* out = (float*)d_out;

  // ---- workspace bump allocator (256B aligned) ----
  char* w = (char*)d_ws;
  size_t off = 0;
  auto alloc = [&](size_t bytes) -> float* {
    float* p = (float*)(w + off);
    off = (off + bytes + 255) & ~(size_t)255;
    return p;
  };
  float* deg    = alloc((size_t)NNODES * 4);
  float* esum   = alloc((size_t)NNODES * 4);
  float* mean_e = alloc((size_t)NNODES * 4);
  float* hA     = alloc((size_t)NNODES * HIDDIM * 4);
  float* hB     = alloc((size_t)NNODES * HIDDIM * 4);
  float* ss     = alloc((size_t)NNODES * 2 * 4);
  float* sd     = alloc((size_t)NNODES * 2 * 4);
  float* mmax   = alloc((size_t)NNODES * 2 * 4);
  float* den    = alloc((size_t)NNODES * 2 * 4);
  float* wa     = alloc(16);
  float* pool   = alloc((size_t)(2 * NGRAPH * HIDDIM + 2 * NGRAPH) * 4);
  float* cnt    = pool + 2 * NGRAPH * HIDDIM;
  float* agg    = alloc((size_t)NNODES * INDIM * 4);     // 51.2 MB arena
  // After the SAGE GEMM, agg is dead: reuse the arena.
  float* hh     = agg;                                   // [N, H*64] <= 25.6 MB
  float* outacc = agg + (size_t)NNODES * 128;            // [N, 64]   12.8 MB
  float* alpha  = agg + (size_t)NNODES * 128 + (size_t)NNODES * 64;  // [ESL, H]

  const int T = 256;
  const float NINF = -__builtin_inff();

  // ---- SAGE: mean-aggregate neighbors, fused dual-GEMM + bias + relu ----
  k_fill<<<cdiv_i(NNODES, T), T, 0, stream>>>(deg,  0.0f, NNODES);
  k_fill<<<cdiv_i(NNODES, T), T, 0, stream>>>(esum, 0.0f, NNODES);
  k_fill<<<cdiv_i((long long)NNODES * INDIM, T), T, 0, stream>>>(agg, 0.0f, (long long)NNODES * INDIM);
  k_deg<<<cdiv_i(NEDGES, T), T, 0, stream>>>(ei, eattr, deg, esum);
  k_scatter_feat<<<cdiv_i((long long)NEDGES * 64, T), T, 0, stream>>>(x, ei, agg);
  k_mean<<<cdiv_i(NNODES * 64, T), T, 0, stream>>>(agg, deg, esum, mean_e);
  k_gemm16<<<dim3(NNODES / 16, HIDDIM / 16), 32, 0, stream>>>(
      agg, sage_wl, x, sage_wr, sage_bl, 1, hA, INDIM, HIDDIM);

  // ---- three GAT layers ----
  struct GatP { const float *gw, *gas, *gad, *gwe, *gae, *gb; int H; };
  GatP layers[3] = {
    { (const float*)d_in[12], (const float*)d_in[13], (const float*)d_in[14],
      (const float*)d_in[15], (const float*)d_in[16], (const float*)d_in[17], 2 },
    { (const float*)d_in[18], (const float*)d_in[19], (const float*)d_in[20],
      (const float*)d_in[21], (const float*)d_in[22], (const float*)d_in[23], 2 },
    { (const float*)d_in[24], (const float*)d_in[25], (const float*)d_in[26],
      (const float*)d_in[27], (const float*)d_in[28], (const float*)d_in[29], 1 },
  };
  float* hin  = hA;
  float* hout = hB;
  for (int l = 0; l < 3; ++l) {
    const GatP& P = layers[l];
    const int H = P.H;
    // hh = h_in @ W^T  (WMMA)
    k_gemm16<<<dim3(NNODES / 16, (H * HIDDIM) / 16), 32, 0, stream>>>(
        hin, P.gw, nullptr, nullptr, nullptr, 0, hh, HIDDIM, H * HIDDIM);
    k_wa<<<1, 32, 0, stream>>>(P.gwe, P.gae, wa, H);
    k_scores<<<cdiv_i((long long)NNODES * H, T), T, 0, stream>>>(hh, P.gas, P.gad, ss, sd, H);
    k_fill<<<cdiv_i((long long)NNODES * H, T), T, 0, stream>>>(mmax, NINF, (long long)NNODES * H);
    k_fill<<<cdiv_i((long long)NNODES * H, T), T, 0, stream>>>(den, 0.0f, (long long)NNODES * H);
    k_fill<<<cdiv_i((long long)NNODES * HIDDIM, T), T, 0, stream>>>(outacc, 0.0f, (long long)NNODES * HIDDIM);
    k_alpha<<<cdiv_i((long long)ESL * H, T), T, 0, stream>>>(
        ei, eattr, mean_e, ss, sd, wa, alpha, mmax, H);
    k_expden<<<cdiv_i((long long)ESL * H, T), T, 0, stream>>>(ei, alpha, mmax, den, H);
    k_aggmsg<<<cdiv_i((long long)ESL * H * 16, T), T, 0, stream>>>(ei, alpha, den, hh, outacc, H);
    k_finish<<<cdiv_i(NNODES * HIDDIM, T), T, 0, stream>>>(outacc, P.gb, hout, 1.0f / H);
    float* t = hin; hin = hout; hout = t;   // ping-pong; final h ends in `hin`
  }

  // ---- masked mean pool (types 1,2) + MLP head ----
  k_fill<<<cdiv_i(2 * NGRAPH * HIDDIM + 2 * NGRAPH, T), T, 0, stream>>>(
      pool, 0.0f, 2 * NGRAPH * HIDDIM + 2 * NGRAPH);
  k_pool<<<cdiv_i(NNODES * 16, T), T, 0, stream>>>(hin, ntype, batch, pool, cnt);
  k_mlp<<<NGRAPH, 64, 0, stream>>>(pool, cnt, mlp_w1, mlp_b1, mlp_w2, mlp_b2, out);
}
